// ExampleModel_1726576856622
// MI455X (gfx1250) — matrix operand order repaired and checked by reading, not compiled, not run on hardware
//
#include <hip/hip_runtime.h>
#include <stdint.h>
#include <stddef.h>

// ---------------- problem constants (from reference) ----------------
constexpr int BB   = 4;
constexpr int SS   = 2048;
constexpr int DDIM = 2048;   // model dim
constexpr int EEXP = 8;      // experts
constexpr int HDIM = 2048;   // hidden
constexpr int TTOK = BB * SS;        // 8192 tokens
constexpr int CCAP = (2 * TTOK) / EEXP; // 2048 capacity

typedef __attribute__((ext_vector_type(16))) __bf16 v16bf;
typedef __attribute__((ext_vector_type(8)))  __bf16 v8bf;
typedef __attribute__((ext_vector_type(8)))  float  v8f;

__device__ __forceinline__ unsigned short f2bf(float f) {
  unsigned u = __float_as_uint(f);
  unsigned r = u + 0x7fffu + ((u >> 16) & 1u);   // round-to-nearest-even
  return (unsigned short)(r >> 16);
}

// ---------------- 1: tokens f32 -> bf16 ----------------
__global__ void __launch_bounds__(256)
conv_tokens(const float* __restrict__ x, unsigned short* __restrict__ o) {
  size_t i = (size_t)blockIdx.x * 256 + threadIdx.x;      // over (T*D)/4
  const float4 v = ((const float4*)x)[i];
  unsigned lo = (unsigned)f2bf(v.x) | ((unsigned)f2bf(v.y) << 16);
  unsigned hi = (unsigned)f2bf(v.z) | ((unsigned)f2bf(v.w) << 16);
  ((uint2*)o)[i] = make_uint2(lo, hi);
}

// ---------------- 2: W1[E,D,H] f32 -> W1t[E,H,D] bf16 (transpose) ----------------
__global__ void __launch_bounds__(256)
transpose_w1(const float* __restrict__ W1, unsigned short* __restrict__ W1t) {
  __shared__ float tl[32][33];
  const int HT = HDIM / 32, DT = DDIM / 32;
  int tile = blockIdx.x;
  int e    = tile / (DT * HT);
  int rem  = tile - e * DT * HT;
  int d0   = (rem / HT) * 32;
  int j0   = (rem % HT) * 32;
  int tx = threadIdx.x, ty = threadIdx.y;
  for (int r = ty; r < 32; r += 8)
    tl[r][tx] = W1[((size_t)e * DDIM + d0 + r) * HDIM + j0 + tx];
  __syncthreads();
  for (int r = ty; r < 32; r += 8)
    W1t[((size_t)e * HDIM + j0 + r) * DDIM + d0 + tx] = f2bf(tl[tx][r]);
}

// ---------------- 3: generic row sum over D (w2sum rows, b2sum rows) ----------------
__global__ void __launch_bounds__(256)
rowsum_kernel(const float* __restrict__ in, float* __restrict__ out) {
  __shared__ float red[256];
  int row = blockIdx.x;
  const float* p = in + (size_t)row * DDIM;
  float sm = 0.f;
  for (int d = threadIdx.x; d < DDIM; d += 256) sm += p[d];
  red[threadIdx.x] = sm; __syncthreads();
  for (int o = 128; o > 0; o >>= 1) {
    if ((int)threadIdx.x < o) red[threadIdx.x] += red[threadIdx.x + o];
    __syncthreads();
  }
  if (threadIdx.x == 0) out[row] = red[0];
}

// ---------------- 4: gating (wave per token) ----------------
__global__ void __launch_bounds__(256)
gate_kernel(const float* __restrict__ x, const float* __restrict__ Wg,
            int* __restrict__ top_idx, float* __restrict__ gates) {
  const int lane = threadIdx.x & 31;
  const int t    = blockIdx.x * 8 + (threadIdx.x >> 5);
  float acc[8] = {0,0,0,0,0,0,0,0};
  const float* xr = x + (size_t)t * DDIM;
  for (int d = lane; d < DDIM; d += 32) {
    float xv = xr[d];
    const float* wg = Wg + d * EEXP;
#pragma unroll
    for (int ee = 0; ee < 8; ++ee) acc[ee] = fmaf(xv, wg[ee], acc[ee]);
  }
#pragma unroll
  for (int off = 16; off >= 1; off >>= 1)
#pragma unroll
    for (int ee = 0; ee < 8; ++ee) acc[ee] += __shfl_xor(acc[ee], off, 32);
  if (lane == 0) {
    float m = acc[0];
#pragma unroll
    for (int ee = 1; ee < 8; ++ee) m = fmaxf(m, acc[ee]);
    float p[8];
#pragma unroll
    for (int ee = 0; ee < 8; ++ee) p[ee] = expf(acc[ee] - m);
    int i0 = 0;
#pragma unroll
    for (int ee = 1; ee < 8; ++ee) if (p[ee] > p[i0]) i0 = ee;   // lowest idx on tie
    int i1 = (i0 == 0) ? 1 : 0;
#pragma unroll
    for (int ee = 0; ee < 8; ++ee) if (ee != i0 && p[ee] > p[i1]) i1 = ee;
    float den = p[i0] + p[i1];
    top_idx[2 * t + 0] = i0;  top_idx[2 * t + 1] = i1;
    gates[2 * t + 0] = p[i0] / den;  gates[2 * t + 1] = p[i1] / den;
  }
}

// ---------------- 5: capacity dispatch (GShard priority: k=0 first, token order) -------
__global__ void __launch_bounds__(256)
dispatch_kernel(const int* __restrict__ top_idx, int* __restrict__ slot_token,
                int* __restrict__ dest) {
  for (int i = threadIdx.x; i < EEXP * CCAP; i += 256) slot_token[i] = -1;
  __syncthreads();
  if ((int)threadIdx.x < EEXP) {
    int e = threadIdx.x, cnt = 0;
    for (int kk = 0; kk < 2; ++kk)
      for (int t = 0; t < TTOK; ++t) {
        if (top_idx[2 * t + kk] == e) {
          if (cnt < CCAP) { slot_token[e * CCAP + cnt] = t; dest[2 * t + kk] = e * CCAP + cnt; }
          else            { dest[2 * t + kk] = -1; }
          ++cnt;
        }
      }
  }
}

// ---------------- 6: fused expert FFN via WMMA, M=32 x N=32 block tile ---------------
// One wave per (expert, 32-slot tile). s[e,c] = relu(x_c W1[e] + b1[e]) . w2sum[e] + b2sum[e]
__global__ void __launch_bounds__(32)
moe_ffn_wmma(const __bf16* __restrict__ tokbf, const __bf16* __restrict__ W1t,
             const float* __restrict__ b1, const float* __restrict__ w2s,
             const float* __restrict__ b2s, const int* __restrict__ slot_token,
             float* __restrict__ s_out) {
  constexpr int ROWP = DDIM + 8;                  // bank-conflict padding (bf16 elems)
  __shared__ __bf16 Atile[32 * ROWP];             // ~128.5 KB of the 320 KB WGP LDS
  __shared__ int ssl[32];

  const int lane = threadIdx.x;
  const int e    = blockIdx.x >> 6;               // C/32 = 64 tiles per expert
  const int c0   = (blockIdx.x & 63) << 5;

  ssl[lane] = slot_token[e * CCAP + c0 + lane];
  __syncthreads();

  // gather 32 token rows (or zeros) into LDS, 16B chunks
  for (int idx = lane; idx < 32 * (DDIM / 8); idx += 32) {
    int row = idx >> 8, ch = idx & 255;
    int t = ssl[row];
    v8bf val = {};
    if (t >= 0) val = *(const v8bf*)(tokbf + (size_t)t * DDIM + ch * 8);
    *(v8bf*)(Atile + row * ROWP + ch * 8) = val;
  }
  __syncthreads();

  const int mrow  = lane & 15;
  const int khalf = lane >> 4;
  // A layout (ISA 7.12.2, 16-bit A 16x32): element i <-> k = (i/8)*16 + khalf*8 + (i%8)
  const __bf16* arow0 = Atile + mrow * ROWP + khalf * 8;          // rows c0+0..15
  const __bf16* arow1 = Atile + (16 + mrow) * ROWP + khalf * 8;   // rows c0+16..31

  float sacc0[8] = {0,0,0,0,0,0,0,0};
  float sacc1[8] = {0,0,0,0,0,0,0,0};

  for (int jt = 0; jt < HDIM / 32; ++jt) {
    const int jA = jt * 32 + (lane & 15);
    const int jB = jA + 16;
    // B layout: element i <-> k = (i/8)*16 + khalf*8 + (i%8), contiguous in d after transpose
    const __bf16* bptrA = W1t + ((size_t)e * HDIM + jA) * DDIM + khalf * 8;
    const __bf16* bptrB = W1t + ((size_t)e * HDIM + jB) * DDIM + khalf * 8;
    v8f acc00 = {}, acc01 = {}, acc10 = {}, acc11 = {};
#pragma unroll 2
    for (int kc = 0; kc < DDIM; kc += 32) {
      v8bf a00 = *(const v8bf*)(arow0 + kc);
      v8bf a01 = *(const v8bf*)(arow0 + kc + 16);
      v8bf a10 = *(const v8bf*)(arow1 + kc);
      v8bf a11 = *(const v8bf*)(arow1 + kc + 16);
      v8bf bA0 = *(const v8bf*)(bptrA + kc);
      v8bf bA1 = *(const v8bf*)(bptrA + kc + 16);
      v8bf bB0 = *(const v8bf*)(bptrB + kc);
      v8bf bB1 = *(const v8bf*)(bptrB + kc + 16);
      v16bf av0 = __builtin_shufflevector(a00, a01, 0,1,2,3,4,5,6,7,8,9,10,11,12,13,14,15);
      v16bf av1 = __builtin_shufflevector(a10, a11, 0,1,2,3,4,5,6,7,8,9,10,11,12,13,14,15);
      v16bf bvA = __builtin_shufflevector(bA0, bA1, 0,1,2,3,4,5,6,7,8,9,10,11,12,13,14,15);
      v16bf bvB = __builtin_shufflevector(bB0, bB1, 0,1,2,3,4,5,6,7,8,9,10,11,12,13,14,15);
      acc00 = __builtin_amdgcn_wmma_f32_16x16x32_bf16(false, av0, false, bvA, (short)0, acc00, false, false);
      acc01 = __builtin_amdgcn_wmma_f32_16x16x32_bf16(false, av0, false, bvB, (short)0, acc01, false, false);
      acc10 = __builtin_amdgcn_wmma_f32_16x16x32_bf16(false, av1, false, bvA, (short)0, acc10, false, false);
      acc11 = __builtin_amdgcn_wmma_f32_16x16x32_bf16(false, av1, false, bvB, (short)0, acc11, false, false);
    }
    // epilogue: relu(+b1) then dot with w2sum column weight, fold into per-row sums
    float bbA = b1[e * HDIM + jA], wwA = w2s[e * HDIM + jA];
    float bbB = b1[e * HDIM + jB], wwB = w2s[e * HDIM + jB];
#pragma unroll
    for (int r = 0; r < 8; ++r) {
      float h;
      h = acc00[r] + bbA; h = h > 0.f ? h : 0.f; sacc0[r] += h * wwA;
      h = acc01[r] + bbB; h = h > 0.f ? h : 0.f; sacc0[r] += h * wwB;
      h = acc10[r] + bbA; h = h > 0.f ? h : 0.f; sacc1[r] += h * wwA;
      h = acc11[r] + bbB; h = h > 0.f ? h : 0.f; sacc1[r] += h * wwB;
    }
  }
  // reduce across the 16 lanes of each half (columns n)
#pragma unroll
  for (int off = 1; off <= 8; off <<= 1)
#pragma unroll
    for (int r = 0; r < 8; ++r) {
      sacc0[r] += __shfl_xor(sacc0[r], off, 32);
      sacc1[r] += __shfl_xor(sacc1[r], off, 32);
    }

  if ((lane & 15) == 0) {
    float badd = b2s[e];
    int mbase = khalf * 8;                        // row m = r + 8*khalf (per 16-row tile)
#pragma unroll
    for (int r = 0; r < 8; ++r) {
      s_out[e * CCAP + c0 + mbase + r]      = sacc0[r] + badd;
      s_out[e * CCAP + c0 + 16 + mbase + r] = sacc1[r] + badd;
    }
  }
}

// ---------------- 7: combine ----------------
__global__ void __launch_bounds__(256)
combine_kernel(const float* __restrict__ gates, const int* __restrict__ dest,
               const float* __restrict__ s, float* __restrict__ ysum) {
  int t = blockIdx.x * 256 + threadIdx.x;
  float a = 0.f;
#pragma unroll
  for (int kk = 0; kk < 2; ++kk) {
    int dd = dest[2 * t + kk];
    if (dd >= 0) a += gates[2 * t + kk] * s[dd];
  }
  ysum[t] = a;
}

// ---------------- 8: log_softmax over S per batch row ----------------
__global__ void __launch_bounds__(256)
lsm_kernel(const float* __restrict__ y, float* __restrict__ out) {
  __shared__ float red[256];
  __shared__ float bval;
  int b = blockIdx.x, tid = threadIdx.x;
  const float* yr = y + (size_t)b * SS;
  float m = -3.4e38f;
  for (int s2 = tid; s2 < SS; s2 += 256) m = fmaxf(m, yr[s2]);
  red[tid] = m; __syncthreads();
  for (int o = 128; o > 0; o >>= 1) {
    if (tid < o) red[tid] = fmaxf(red[tid], red[tid + o]);
    __syncthreads();
  }
  if (tid == 0) bval = red[0];
  __syncthreads();
  float mx = bval, sm = 0.f;
  for (int s2 = tid; s2 < SS; s2 += 256) sm += expf(yr[s2] - mx);
  __syncthreads();
  red[tid] = sm; __syncthreads();
  for (int o = 128; o > 0; o >>= 1) {
    if (tid < o) red[tid] += red[tid + o];
    __syncthreads();
  }
  if (tid == 0) bval = mx + logf(red[0]);
  __syncthreads();
  float lg = bval;
  for (int s2 = tid; s2 < SS; s2 += 256) out[(size_t)b * SS + s2] = yr[s2] - lg;
}

// ---------------- launch ----------------
extern "C" void kernel_launch(void* const* d_in, const int* in_sizes, int n_in,
                              void* d_out, int out_size, void* d_ws, size_t ws_size,
                              hipStream_t stream) {
  const float* x  = (const float*)d_in[0];   // [B,S,D]
  const float* Wg = (const float*)d_in[1];   // [D,E]
  const float* W1 = (const float*)d_in[2];   // [E,D,H]
  const float* b1 = (const float*)d_in[3];   // [E,H]
  const float* W2 = (const float*)d_in[4];   // [E,H,D]
  const float* b2 = (const float*)d_in[5];   // [E,D]
  float* out = (float*)d_out;                // [B,S]

  auto up = [](size_t v) { return (v + 255) & ~(size_t)255; };
  char* w = (char*)d_ws;
  size_t off = 0;
  unsigned short* tokbf = (unsigned short*)(w + off); off += up((size_t)TTOK * DDIM * 2);
  unsigned short* W1t   = (unsigned short*)(w + off); off += up((size_t)EEXP * HDIM * DDIM * 2);
  float* w2sum = (float*)(w + off); off += up((size_t)EEXP * HDIM * 4);
  float* b2sum = (float*)(w + off); off += up((size_t)EEXP * 4);
  int*   tidx  = (int*)  (w + off); off += up((size_t)TTOK * 2 * 4);
  float* gts   = (float*)(w + off); off += up((size_t)TTOK * 2 * 4);
  int*   slot  = (int*)  (w + off); off += up((size_t)EEXP * CCAP * 4);
  int*   dst   = (int*)  (w + off); off += up((size_t)TTOK * 2 * 4);
  float* svec  = (float*)(w + off); off += up((size_t)EEXP * CCAP * 4);
  float* ysum  = (float*)(w + off); off += up((size_t)TTOK * 4);

  conv_tokens <<<(TTOK * DDIM) / (256 * 4), 256, 0, stream>>>(x, tokbf);
  transpose_w1<<<EEXP * (DDIM / 32) * (HDIM / 32), dim3(32, 8), 0, stream>>>(W1, W1t);
  rowsum_kernel<<<EEXP * HDIM, 256, 0, stream>>>(W2, w2sum);
  rowsum_kernel<<<EEXP, 256, 0, stream>>>(b2, b2sum);
  gate_kernel <<<TTOK / 8, 256, 0, stream>>>(x, Wg, tidx, gts);
  dispatch_kernel<<<1, 256, 0, stream>>>(tidx, slot, dst);
  moe_ffn_wmma<<<EEXP * (CCAP / 32), 32, 0, stream>>>(
      (const __bf16*)tokbf, (const __bf16*)W1t, b1, w2sum, b2sum, slot, svec);
  combine_kernel<<<TTOK / 256, 256, 0, stream>>>(gts, dst, svec, ysum);
  lsm_kernel<<<BB, 256, 0, stream>>>(ysum, out);
}
